// TFDAT_11012296147318
// MI455X (gfx1250) — matrix operand order repaired and checked by reading, not compiled, not run on hardware
//
#include <hip/hip_runtime.h>

#define B0 2
#define T0 5
#define H0 256
#define W0 256
#define L0 (H0*W0)
#define D 64
#define NH 4
#define HD 16
#define FFH 128
#define MID 64
#define NGROUPS 4
#define DPG 3

typedef __attribute__((ext_vector_type(16))) __bf16 v16bf;
typedef __attribute__((ext_vector_type(8)))  float  v8f;

union FragA { v16bf v; uint4 q[2]; };

__device__ __forceinline__ unsigned short f2bf(float f){
  union { float f; unsigned u; } c; c.f = f;
  unsigned u = c.u + 0x7fffu + ((c.u >> 16) & 1u);
  return (unsigned short)(u >> 16);
}
__device__ __forceinline__ unsigned pk2(float a, float b){
  return (unsigned)f2bf(a) | ((unsigned)f2bf(b) << 16);
}
__device__ __forceinline__ float geluf(float x){ return 0.5f*x*(1.0f+erff(x*0.70710678118f)); }
__device__ __forceinline__ float sigm(float x){ return 1.0f/(1.0f+__expf(-x)); }

// xor-shuffle within 16-lane rows, VALU-only via v_permlane16_b32
template<int O>
__device__ __forceinline__ float xshfl16(float v){
#if __has_builtin(__builtin_amdgcn_permlane16)
  unsigned s0, s1;
  if (O == 1)      { s0 = 0x67452301u; s1 = 0xefcdab89u; }
  else if (O == 2) { s0 = 0x54761032u; s1 = 0xdcfe98bau; }
  else if (O == 4) { s0 = 0x32107654u; s1 = 0xba98fedcu; }
  else             { s0 = 0xfedcba98u; s1 = 0x76543210u; }
  unsigned u = __float_as_uint(v);
  return __uint_as_float(__builtin_amdgcn_permlane16(u, u, s0, s1, false, false));
#else
  return __shfl_xor(v, O, 16);
#endif
}
__device__ __forceinline__ float rowmax16(float v){
  v = fmaxf(v, xshfl16<8>(v));
  v = fmaxf(v, xshfl16<4>(v));
  v = fmaxf(v, xshfl16<2>(v));
  v = fmaxf(v, xshfl16<1>(v));
  return v;
}
__device__ __forceinline__ float rowsum16(float v){
  v += xshfl16<8>(v);
  v += xshfl16<4>(v);
  v += xshfl16<2>(v);
  v += xshfl16<1>(v);
  return v;
}

// ---------------------------------------------------------------------------
// WMMA bf16 GEMM / implicit-GEMM conv-tap kernel, specialized by template:
//   CONVM: 0 = A row-major [M][K]; 1 = spatial-shift gather (3x3 conv tap)
//   ANCHW: conv source layout NCHW (1) vs token-major (0)
// Block: 256 threads = 8 waves; tile M=256, N=64, K-step 32.
// ---------------------------------------------------------------------------
#define GBM 256
#define GBN 64
#define GBK 32
#define GAST 40   // padded LDS row stride (ushorts); 80B keeps 16B alignment

template<int CONVM, int ANCHW>
__global__ void __launch_bounds__(256)
k_gemm_t(const float* __restrict__ A, const unsigned short* __restrict__ Wb,
         const float* __restrict__ bias, const float* __restrict__ resid,
         float* __restrict__ C, int M, int N, int K,
         int act, int accum, int Hc, int Wc, int dy, int dx, int out_nchw)
{
  __shared__ unsigned short As[GBM * GAST];
  __shared__ unsigned short Bs[GBN * GAST];
  const int tid  = threadIdx.x;
  const int lane = tid & 31, wave = tid >> 5;
  const int m0 = blockIdx.y * GBM, n0 = blockIdx.x * GBN;
  const int HWc = Hc * Wc;

  // per-thread A source row (row index within tile == tid); mask once.
  const float* arow = nullptr;
  bool arow_ok = true;
  {
    int m = m0 + tid;                       // M is always a multiple of GBM here
    if (CONVM) {
      int b = m / HWc, pix = m - b * HWc;
      int y = pix / Wc + dy, x = pix % Wc + dx;
      arow_ok = (y >= 0 && y < Hc && x >= 0 && x < Wc);
      if (ANCHW) arow = A + (size_t)b * K * HWc + (size_t)y * Wc + x;   // +k*HWc
      else       arow = A + ((size_t)b * HWc + (size_t)y * Wc + x) * (size_t)K;
    } else {
      arow = A + (size_t)m * K;
    }
  }

  v8f zero8 = {0,0,0,0,0,0,0,0};
  v8f acc[2][4];
#pragma unroll
  for (int mi = 0; mi < 2; ++mi)
#pragma unroll
    for (int ni = 0; ni < 4; ++ni) acc[mi][ni] = zero8;

  for (int k0 = 0; k0 < K; k0 += GBK) {
    // ---- stage A: one full row per thread ----
    unsigned short* dstA = &As[tid * GAST];
    if (CONVM && ANCHW) {
#pragma unroll
      for (int kk = 0; kk < GBK; ++kk) {
        float v = (arow_ok && (k0 + kk) < K) ? arow[(size_t)(k0 + kk) * HWc] : 0.f;
        dstA[kk] = f2bf(v);
      }
    } else {
      if (arow_ok && (K - k0) >= GBK) {
        const float4* src = (const float4*)(arow + k0);
#pragma unroll
        for (int e = 0; e < 4; ++e) {
          float4 a = src[2*e], b = src[2*e + 1];
          *(uint4*)(dstA + e*8) = make_uint4(pk2(a.x,a.y), pk2(a.z,a.w),
                                             pk2(b.x,b.y), pk2(b.z,b.w));
        }
      } else {
#pragma unroll
        for (int kk = 0; kk < GBK; ++kk) {
          float v = (arow_ok && (k0 + kk) < K) ? arow[k0 + kk] : 0.f;
          dstA[kk] = f2bf(v);
        }
      }
    }
    // ---- stage B transposed: Bs[n][kk] = W[k0+kk][n0+n] ----
#pragma unroll
    for (int e = 0; e < 8; ++e) {
      int idx = tid + e * 256;             // 0..2047
      int kk = idx >> 6, n = idx & 63;
      unsigned short v = 0;
      if (k0 + kk < K && n0 + n < N) v = Wb[(size_t)(k0 + kk) * N + (n0 + n)];
      Bs[n * GAST + kk] = v;
    }
    if (k0 + GBK < K)
      __builtin_prefetch(Wb + (size_t)(k0 + GBK) * N + n0 + (tid & 63), 0, 0);
    __syncthreads();

    // ---- fragments per CDNA5 16-bit A / B layouts ----
    const int ar  = wave * 32 + (lane & 15);
    const int akb = (lane < 16) ? 0 : 8;    // A: lo=K kb..kb+7, hi=K kb+16..kb+23
    const int bc  = lane & 15;
    const int bkb = (lane < 16) ? 0 : 8;    // B: lo=K kb..kb+7, hi=K kb+16..kb+23
    FragA af[2], bfr[4];
#pragma unroll
    for (int mi = 0; mi < 2; ++mi) {
      const unsigned short* p = &As[(ar + mi * 16) * GAST + akb];
      af[mi].q[0] = *(const uint4*)p;
      af[mi].q[1] = *(const uint4*)(p + 16);
    }
#pragma unroll
    for (int ni = 0; ni < 4; ++ni) {
      const unsigned short* p = &Bs[(bc + ni * 16) * GAST + bkb];
      bfr[ni].q[0] = *(const uint4*)p;
      bfr[ni].q[1] = *(const uint4*)(p + 16);
    }
#pragma unroll
    for (int mi = 0; mi < 2; ++mi)
#pragma unroll
      for (int ni = 0; ni < 4; ++ni)
        acc[mi][ni] = __builtin_amdgcn_wmma_f32_16x16x32_bf16(
            false, af[mi].v, false, bfr[ni].v, (short)0, acc[mi][ni], false, false);
    __syncthreads();
  }

  // ---- epilogue (C/D layout: VGPR r -> M = r + (lane<16?0:8)) ----
#pragma unroll
  for (int mi = 0; mi < 2; ++mi)
#pragma unroll
    for (int ni = 0; ni < 4; ++ni)
#pragma unroll
      for (int r = 0; r < 8; ++r) {
        int m = m0 + wave * 32 + mi * 16 + r + ((lane < 16) ? 0 : 8);
        int n = n0 + ni * 16 + (lane & 15);
        if (m < M && n < N) {
          size_t oidx;
          if (out_nchw) { int b = m / HWc, pix = m - b * HWc;
                          oidx = ((size_t)b * N + n) * HWc + pix; }
          else oidx = (size_t)m * N + n;
          float v = acc[mi][ni][r];
          if (accum) v += C[oidx];
          if (bias)  v += bias[n];
          if (act == 1) v = geluf(v);
          else if (act == 2) v = (v > 0.f) ? v : 0.2f * v;
          if (resid) v += resid[oidx];
          C[oidx] = v;
        }
      }
}

// ---------------------------------------------------------------------------
// Window attention: 1 block per 8x8 window, wave h = head h (wave32).
// Ks is zero-padded to K=32 so all 32 lanes load B fragments unconditionally.
// Softmax row reductions use v_permlane16 (VALU) instead of LDS bpermute.
// ---------------------------------------------------------------------------
__global__ void __launch_bounds__(128)
k_win_attn(const float* __restrict__ qkv, const float* __restrict__ relb,
           float* __restrict__ outp)
{
  __shared__ unsigned short Qs[NH][64][16];
  __shared__ unsigned short Ks[NH][64][32];   // [.. ][16..31] stays zero
  __shared__ unsigned short Vt[NH][16][72];
  __shared__ unsigned short Pb[NH][16][72];
  const int nw = W0 / 8;
  int blk = blockIdx.x;
  int b = blk / (nw * nw);
  int wrem = blk - b * nw * nw;
  int wy = wrem / nw, wx = wrem - wy * nw;
  int lane = threadIdx.x & 31, h = threadIdx.x >> 5;

  const uint4 uz = make_uint4(0, 0, 0, 0);
#pragma unroll
  for (int ii = 0; ii < 2; ++ii) {
    int i = lane + ii * 32;
    int iy = i >> 3, ix = i & 7;
    size_t tok = (size_t)b * L0 + (size_t)(wy * 8 + iy) * W0 + (wx * 8 + ix);
    const float* p = qkv + tok * (3 * D);
    const float4* q4 = (const float4*)(p + h * 16);
    const float4* k4 = (const float4*)(p + 64 + h * 16);
    const float4* v4 = (const float4*)(p + 128 + h * 16);
    float4 q0 = q4[0], q1 = q4[1], q2 = q4[2], q3 = q4[3];
    *(uint4*)&Qs[h][i][0] = make_uint4(pk2(q0.x,q0.y), pk2(q0.z,q0.w),
                                       pk2(q1.x,q1.y), pk2(q1.z,q1.w));
    *(uint4*)&Qs[h][i][8] = make_uint4(pk2(q2.x,q2.y), pk2(q2.z,q2.w),
                                       pk2(q3.x,q3.y), pk2(q3.z,q3.w));
    float4 k0 = k4[0], k1 = k4[1], k2 = k4[2], k3 = k4[3];
    *(uint4*)&Ks[h][i][0]  = make_uint4(pk2(k0.x,k0.y), pk2(k0.z,k0.w),
                                        pk2(k1.x,k1.y), pk2(k1.z,k1.w));
    *(uint4*)&Ks[h][i][8]  = make_uint4(pk2(k2.x,k2.y), pk2(k2.z,k2.w),
                                        pk2(k3.x,k3.y), pk2(k3.z,k3.w));
    *(uint4*)&Ks[h][i][16] = uz;
    *(uint4*)&Ks[h][i][24] = uz;
    float4 v0 = v4[0], v1 = v4[1], v2 = v4[2], v3 = v4[3];
    Vt[h][ 0][i] = f2bf(v0.x); Vt[h][ 1][i] = f2bf(v0.y);
    Vt[h][ 2][i] = f2bf(v0.z); Vt[h][ 3][i] = f2bf(v0.w);
    Vt[h][ 4][i] = f2bf(v1.x); Vt[h][ 5][i] = f2bf(v1.y);
    Vt[h][ 6][i] = f2bf(v1.z); Vt[h][ 7][i] = f2bf(v1.w);
    Vt[h][ 8][i] = f2bf(v2.x); Vt[h][ 9][i] = f2bf(v2.y);
    Vt[h][10][i] = f2bf(v2.z); Vt[h][11][i] = f2bf(v2.w);
    Vt[h][12][i] = f2bf(v3.x); Vt[h][13][i] = f2bf(v3.y);
    Vt[h][14][i] = f2bf(v3.z); Vt[h][15][i] = f2bf(v3.w);
  }
  __syncthreads();

#pragma unroll 1
  for (int qi = 0; qi < 4; ++qi) {
    FragA af;
    int rowA = qi * 16 + (lane & 15);
    int akb = (lane < 16) ? 0 : 8;
    af.q[0] = *(const uint4*)&Qs[h][rowA][akb];
    af.q[1] = uz;                              // K=16..31 zero-padded
    v8f s[4];
#pragma unroll
    for (int ki = 0; ki < 4; ++ki) {
      FragA bfr;
      const unsigned short* p = &Ks[h][ki * 16 + (lane & 15)][(lane < 16) ? 0 : 8];
      bfr.q[0] = *(const uint4*)p;
      bfr.q[1] = *(const uint4*)(p + 16);
      v8f z = {0,0,0,0,0,0,0,0};
      s[ki] = __builtin_amdgcn_wmma_f32_16x16x32_bf16(false, af.v, false, bfr.v,
                                                      (short)0, z, false, false);
    }
    int qbase = qi * 16 + ((lane < 16) ? 0 : 8);
#pragma unroll
    for (int ki = 0; ki < 4; ++ki)
#pragma unroll
      for (int r = 0; r < 8; ++r)
        s[ki][r] = s[ki][r] * 0.25f +
                   relb[((size_t)h * 64 + (qbase + r)) * 64 + (ki * 16 + (lane & 15))];
    // softmax: each query row = 16 lanes of one half-wave x 4 ki fragments
#pragma unroll
    for (int r = 0; r < 8; ++r) {
      float mx = s[0][r];
#pragma unroll
      for (int ki = 1; ki < 4; ++ki) mx = fmaxf(mx, s[ki][r]);
      mx = rowmax16(mx);
      float sum = 0.f;
#pragma unroll
      for (int ki = 0; ki < 4; ++ki) { float e = __expf(s[ki][r] - mx); s[ki][r] = e; sum += e; }
      sum = rowsum16(sum);
      float inv = 1.f / sum;
      int qr = qbase + r - qi * 16;
#pragma unroll
      for (int ki = 0; ki < 4; ++ki)
        Pb[h][qr][ki * 16 + (lane & 15)] = f2bf(s[ki][r] * inv);
    }
    __syncthreads();
    // O = P(16x64) * V(64x16), two K=32 WMMA steps
    v8f o8 = {0,0,0,0,0,0,0,0};
#pragma unroll
    for (int ks = 0; ks < 2; ++ks) {
      FragA pa, vb;
      int kb = ks * 32 + ((lane < 16) ? 0 : 8);
      const unsigned short* pp = &Pb[h][lane & 15][kb];
      pa.q[0] = *(const uint4*)pp; pa.q[1] = *(const uint4*)(pp + 16);
      int kb2 = ks * 32 + ((lane < 16) ? 0 : 8);
      const unsigned short* pv = &Vt[h][lane & 15][kb2];
      vb.q[0] = *(const uint4*)pv; vb.q[1] = *(const uint4*)(pv + 16);
      o8 = __builtin_amdgcn_wmma_f32_16x16x32_bf16(false, pa.v, false, vb.v,
                                                   (short)0, o8, false, false);
    }
#pragma unroll
    for (int r = 0; r < 8; ++r) {
      int q = qbase + r;
      int iy = q >> 3, ix = q & 7;
      size_t tok = (size_t)b * L0 + (size_t)(wy * 8 + iy) * W0 + (wx * 8 + ix);
      outp[tok * D + h * 16 + (lane & 15)] = o8[r];
    }
    __syncthreads();
  }
}

// ---------------------------------------------------------------------------
// Elementwise / glue kernels (f32, float4-vectorized)
// ---------------------------------------------------------------------------
__global__ void k_layernorm(const float* __restrict__ x, const float* __restrict__ g,
                            const float* __restrict__ bt, float* __restrict__ o, int M)
{
  int m = blockIdx.x * blockDim.x + threadIdx.x;
  if (m >= M) return;
  const float4* p4 = (const float4*)(x + (size_t)m * D);
  float s = 0.f, s2 = 0.f;
#pragma unroll
  for (int c = 0; c < D/4; ++c) {
    float4 v = p4[c];
    s  += v.x + v.y + v.z + v.w;
    s2 += v.x*v.x + v.y*v.y + v.z*v.z + v.w*v.w;
  }
  float mean = s / D, var = s2 / D - mean * mean;
  float r = rsqrtf(var + 1e-5f);
  float4* q4 = (float4*)(o + (size_t)m * D);
  const float4* g4 = (const float4*)g;
  const float4* b4 = (const float4*)bt;
#pragma unroll
  for (int c = 0; c < D/4; ++c) {
    float4 v = p4[c], gg = g4[c], bb = b4[c];
    float4 w;
    w.x = (v.x - mean) * r * gg.x + bb.x;
    w.y = (v.y - mean) * r * gg.y + bb.y;
    w.z = (v.z - mean) * r * gg.z + bb.z;
    w.w = (v.w - mean) * r * gg.w + bb.w;
    q4[c] = w;
  }
}

// depthwise 3x3 conv, token-major, weights pre-transposed to [9][C]; 4 ch/thread
__global__ void k_dwconv4(const float* __restrict__ x, const float* __restrict__ wT,
                          float* __restrict__ o, int C, int actGelu)
{
  size_t i = blockIdx.x * (size_t)blockDim.x + threadIdx.x;
  int C4 = C >> 2;
  size_t total = (size_t)B0 * L0 * C4;
  if (i >= total) return;
  int c4 = (int)(i % C4); size_t t = i / C4;
  int pix = (int)(t % L0); int b = (int)(t / L0);
  int y = pix / W0, xx = pix % W0;
  float4 acc = {0.f, 0.f, 0.f, 0.f};
#pragma unroll
  for (int ty = 0; ty < 3; ++ty) {
    int yy = y + ty - 1; if (yy < 0 || yy >= H0) continue;
#pragma unroll
    for (int tx = 0; tx < 3; ++tx) {
      int xc = xx + tx - 1; if (xc < 0 || xc >= W0) continue;
      float4 xv = *(const float4*)&x[((size_t)b * L0 + (size_t)yy * W0 + xc) * C + c4*4];
      float4 wv = *(const float4*)&wT[(ty*3 + tx) * C + c4*4];
      acc.x += xv.x * wv.x; acc.y += xv.y * wv.y;
      acc.z += xv.z * wv.z; acc.w += xv.w * wv.w;
    }
  }
  if (actGelu) { acc.x = geluf(acc.x); acc.y = geluf(acc.y);
                 acc.z = geluf(acc.z); acc.w = geluf(acc.w); }
  ((float4*)o)[i] = acc;
}

__global__ void k_gap(const float* __restrict__ cf, float* __restrict__ gap)
{
  int b = blockIdx.x / D, c = blockIdx.x % D;
  float s = 0.f;
  for (int i = threadIdx.x; i < L0; i += blockDim.x)
    s += cf[((size_t)b * L0 + i) * D + c];
  __shared__ float red[256];
  red[threadIdx.x] = s; __syncthreads();
  for (int st = 128; st > 0; st >>= 1) {
    if ((int)threadIdx.x < st) red[threadIdx.x] += red[threadIdx.x + st];
    __syncthreads();
  }
  if (threadIdx.x == 0) gap[b * D + c] = red[0] / (float)L0;
}

__global__ void k_cm(const float* __restrict__ gap, const float* __restrict__ w1,
                     const float* __restrict__ w2, float* __restrict__ cm)
{
  int b = blockIdx.x;
  __shared__ float t8[8];
  if (threadIdx.x < 8) {
    float s = 0.f;
    for (int c = 0; c < D; ++c) s += gap[b * D + c] * w1[threadIdx.x * D + c];
    t8[threadIdx.x] = geluf(s);
  }
  __syncthreads();
  if ((int)threadIdx.x < D) {
    float s = 0.f;
    for (int j = 0; j < 8; ++j) s += t8[j] * w2[threadIdx.x * 8 + j];
    cm[b * D + threadIdx.x] = sigm(s);
  }
}

__global__ void k_spat_fuse(float* __restrict__ x, const float* __restrict__ a,
                            const float* __restrict__ cf, const float* __restrict__ cm)
{
  size_t i = blockIdx.x * (size_t)blockDim.x + threadIdx.x;
  size_t total = (size_t)B0 * L0 * (D/4);
  if (i >= total) return;
  int c4 = (int)(i % (D/4)); int b = (int)(i / ((size_t)L0 * (D/4)));
  float4 xv = ((float4*)x)[i], av = ((const float4*)a)[i], cv = ((const float4*)cf)[i];
  float4 mv = *(const float4*)&cm[b * D + c4*4];
  xv.x += av.x * mv.x + cv.x; xv.y += av.y * mv.y + cv.y;
  xv.z += av.z * mv.z + cv.z; xv.w += av.w * mv.w + cv.w;
  ((float4*)x)[i] = xv;
}

__global__ void k_chan_norms(const float* __restrict__ qkv, float* __restrict__ nq,
                             float* __restrict__ nk)
{
  int id = blockIdx.x;                 // (b*NH+h)*16 + d
  int d = id % 16; int h = (id / 16) % NH; int b = id / (16 * NH);
  float sq = 0.f, sk = 0.f;
  for (int n = threadIdx.x; n < L0; n += blockDim.x) {
    const float* p = qkv + ((size_t)b * L0 + n) * (3 * D);
    float q = p[h * 16 + d], k = p[64 + h * 16 + d];
    sq += q * q; sk += k * k;
  }
  __shared__ float r1[256], r2[256];
  r1[threadIdx.x] = sq; r2[threadIdx.x] = sk; __syncthreads();
  for (int st = 128; st > 0; st >>= 1) {
    if ((int)threadIdx.x < st) { r1[threadIdx.x] += r1[threadIdx.x + st];
                                 r2[threadIdx.x] += r2[threadIdx.x + st]; }
    __syncthreads();
  }
  if (threadIdx.x == 0) {
    nq[id] = fmaxf(sqrtf(r1[0]), 1e-12f);
    nk[id] = fmaxf(sqrtf(r2[0]), 1e-12f);
  }
}

__global__ void k_zero(float* p, int n)
{ int i = blockIdx.x * blockDim.x + threadIdx.x; if (i < n) p[i] = 0.f; }

__global__ void k_chan_gram(const float* __restrict__ qkv, float* __restrict__ G)
{
  int seg = blockIdx.x % 64; int bh = blockIdx.x / 64;
  int h = bh % NH, b = bh / NH;
  __shared__ float Qs[128][17], Ks2[128][17];
  int tid = threadIdx.x;
  int d = tid >> 4, e = tid & 15;
  float acc = 0.f;
  for (int c = 0; c < 8; ++c) {
    int n0 = (seg * 8 + c) * 128;
    for (int i = tid; i < 128 * 16; i += 256) {
      int rn = i >> 4, dd = i & 15;
      const float* p = qkv + ((size_t)b * L0 + n0 + rn) * (3 * D);
      Qs[rn][dd]  = p[h * 16 + dd];
      Ks2[rn][dd] = p[64 + h * 16 + dd];
    }
    __syncthreads();
    for (int n = 0; n < 128; ++n) acc += Qs[n][d] * Ks2[n][e];
    __syncthreads();
  }
  atomicAdd(&G[(size_t)bh * 256 + d * 16 + e], acc);
}

__global__ void k_chan_softmax(const float* __restrict__ G, const float* __restrict__ nq,
                               const float* __restrict__ nk, const float* __restrict__ temp,
                               float* __restrict__ attn)
{
  int bh = blockIdx.x; int h = bh % NH;
  int d = threadIdx.x;
  if (d < 16) {
    float row[16]; float mx = -1e30f;
#pragma unroll
    for (int e = 0; e < 16; ++e) {
      float v = G[bh * 256 + d * 16 + e] / (nq[bh * 16 + d] * nk[bh * 16 + e]) * temp[h];
      row[e] = v; mx = fmaxf(mx, v);
    }
    float s = 0.f;
#pragma unroll
    for (int e = 0; e < 16; ++e) { row[e] = __expf(row[e] - mx); s += row[e]; }
#pragma unroll
    for (int e = 0; e < 16; ++e) attn[bh * 256 + d * 16 + e] = row[e] / s;
  }
}

__global__ void k_chan_apply(const float* __restrict__ qkv, const float* __restrict__ attn,
                             float* __restrict__ o)
{
  size_t i = blockIdx.x * (size_t)blockDim.x + threadIdx.x;
  size_t total = (size_t)B0 * L0 * NH;
  if (i >= total) return;
  int h = (int)(i % NH); size_t t = i / NH; int b = (int)(t / L0);
  const float4* v4 = (const float4*)(qkv + t * (3 * D) + 128 + h * 16);
  float4 v0 = v4[0], v1 = v4[1], v2 = v4[2], v3 = v4[3];
  float vv[16] = { v0.x,v0.y,v0.z,v0.w, v1.x,v1.y,v1.z,v1.w,
                   v2.x,v2.y,v2.z,v2.w, v3.x,v3.y,v3.z,v3.w };
  const float* A = attn + ((size_t)b * NH + h) * 256;
  float* op = o + t * D + h * 16;
#pragma unroll
  for (int d = 0; d < 16; ++d) {
    float s = 0.f;
#pragma unroll
    for (int e = 0; e < 16; ++e) s += A[d * 16 + e] * vv[e];
    op[d] = s;
  }
}

__global__ void k_chan_fuse(float* __restrict__ x, const float* __restrict__ a,
                            const float* __restrict__ cf, const float* __restrict__ sg, int M)
{
  int m = blockIdx.x * blockDim.x + threadIdx.x;
  if (m >= M) return;
  const float4* a4 = (const float4*)(a + (size_t)m * D);
  const float4* s4 = (const float4*)sg;
  float s = 0.f;
#pragma unroll
  for (int c = 0; c < D/4; ++c) {
    float4 av = a4[c], sv = s4[c];
    s += av.x*sv.x + av.y*sv.y + av.z*sv.z + av.w*sv.w;
  }
  float sm = sigm(s);
  float4* x4 = (float4*)(x + (size_t)m * D);
  const float4* c4 = (const float4*)(cf + (size_t)m * D);
#pragma unroll
  for (int c = 0; c < D/4; ++c) {
    float4 xv = x4[c], av = a4[c], cv = c4[c];
    xv.x += av.x + cv.x * sm; xv.y += av.y + cv.y * sm;
    xv.z += av.z + cv.z * sm; xv.w += av.w + cv.w * sm;
    x4[c] = xv;
  }
}

// ---- flownet / warp / resize ----
__global__ void k_concat_src_tgt(const float* __restrict__ X, int t, float* __restrict__ z6)
{
  size_t i = blockIdx.x * (size_t)blockDim.x + threadIdx.x;
  size_t total = (size_t)B0 * 6 * L0;
  if (i >= total) return;
  int pix = (int)(i % L0); int c = (int)((i / L0) % 6); int b = (int)(i / (6 * (size_t)L0));
  int tt = (c < 3) ? t : (T0 / 2); int cc = c % 3;
  z6[i] = X[(((size_t)b * T0 + tt) * 3 + cc) * L0 + pix];
}

__global__ void k_conv_direct(const float* __restrict__ in, const float* __restrict__ w,
                              const float* __restrict__ bias, float* __restrict__ out,
                              int Ci, int Co, int Hi, int Wi, int stride, int act)
{
  int Ho = Hi / stride, Wo = Wi / stride;
  size_t i = blockIdx.x * (size_t)blockDim.x + threadIdx.x;
  size_t total = (size_t)B0 * Co * Ho * Wo;
  if (i >= total) return;
  int xo = (int)(i % Wo); int yo = (int)((i / Wo) % Ho);
  int co = (int)((i / ((size_t)Wo * Ho)) % Co); int b = (int)(i / ((size_t)Wo * Ho * Co));
  float acc = bias ? bias[co] : 0.f;
  for (int ci = 0; ci < Ci; ++ci)
    for (int ty = 0; ty < 3; ++ty) {
      int iy = yo * stride + ty - 1; if (iy < 0 || iy >= Hi) continue;
      for (int tx = 0; tx < 3; ++tx) {
        int ix = xo * stride + tx - 1; if (ix < 0 || ix >= Wi) continue;
        acc += in[((size_t)b * Ci + ci) * Hi * Wi + (size_t)iy * Wi + ix] *
               w[((co * Ci + ci) * 3 + ty) * 3 + tx];
      }
    }
  if (act == 2) acc = (acc > 0.f) ? acc : 0.2f * acc;
  out[((size_t)b * Co + co) * Ho * Wo + (size_t)yo * Wo + xo] = acc;
}

__global__ void k_flow_up(const float* __restrict__ fl, float* __restrict__ flow)
{
  size_t i = blockIdx.x * (size_t)blockDim.x + threadIdx.x;
  size_t total = (size_t)B0 * 2 * L0;
  if (i >= total) return;
  int xo = (int)(i % W0); int yo = (int)((i / W0) % H0);
  int c = (int)((i / (size_t)L0) % 2); int b = (int)(i / (2 * (size_t)L0));
  float sx = (xo + 0.5f) * 0.125f - 0.5f;
  float sy = (yo + 0.5f) * 0.125f - 0.5f;
  sx = fminf(fmaxf(sx, 0.f), 31.f); sy = fminf(fmaxf(sy, 0.f), 31.f);
  int x0 = (int)sx, y0 = (int)sy;
  int x1 = min(x0 + 1, 31), y1 = min(y0 + 1, 31);
  float dx = sx - x0, dyf = sy - y0;
  const float* p = fl + ((size_t)b * 2 + c) * 1024;
  float v = (1 - dx) * (1 - dyf) * p[y0 * 32 + x0] + dx * (1 - dyf) * p[y0 * 32 + x1] +
            (1 - dx) * dyf * p[y1 * 32 + x0] + dx * dyf * p[y1 * 32 + x1];
  flow[i] = 8.f * v;
}

__global__ void k_warp(const float* __restrict__ X, int t, const float* __restrict__ flow,
                       float* __restrict__ aligned)
{
  size_t i = blockIdx.x * (size_t)blockDim.x + threadIdx.x;
  size_t total = (size_t)B0 * 3 * L0;
  if (i >= total) return;
  int pix = (int)(i % L0); int c = (int)((i / L0) % 3); int b = (int)(i / (3 * (size_t)L0));
  int y = pix / W0, x = pix % W0;
  float fx = flow[((size_t)b * 2 + 0) * L0 + pix];
  float fy = flow[((size_t)b * 2 + 1) * L0 + pix];
  float nx = fminf(fmaxf((float)x + fx, 0.f), (float)(W0 - 1));
  float ny = fminf(fmaxf((float)y + fy, 0.f), (float)(H0 - 1));
  float x0f = floorf(nx), y0f = floorf(ny);
  int x0 = (int)x0f, y0 = (int)y0f;
  int x1 = min(x0 + 1, W0 - 1), y1 = min(y0 + 1, H0 - 1);
  float dx = nx - x0f, dyw = ny - y0f;
  const float* src = X + (((size_t)b * T0 + t) * 3 + c) * L0;
  float p00 = src[(size_t)y0 * W0 + x0], p01 = src[(size_t)y0 * W0 + x1];
  float p10 = src[(size_t)y1 * W0 + x0], p11 = src[(size_t)y1 * W0 + x1];
  aligned[((size_t)b * 15 + (t * 3 + c)) * L0 + pix] =
      (1 - dx) * (1 - dyw) * p00 + dx * (1 - dyw) * p01 + (1 - dx) * dyw * p10 + dx * dyw * p11;
}

__global__ void k_center_copy(const float* __restrict__ X, float* __restrict__ aligned)
{
  size_t i = blockIdx.x * (size_t)blockDim.x + threadIdx.x;
  size_t total = (size_t)B0 * 3 * L0;
  if (i >= total) return;
  int pix = (int)(i % L0); int c = (int)((i / L0) % 3); int b = (int)(i / (3 * (size_t)L0));
  aligned[((size_t)b * 15 + (6 + c)) * L0 + pix] =
      X[(((size_t)b * T0 + 2) * 3 + c) * L0 + pix];
}

__global__ void k_pixshuf(const float* __restrict__ hu, float* __restrict__ hs)
{
  size_t i = blockIdx.x * (size_t)blockDim.x + threadIdx.x;
  size_t total = (size_t)B0 * 512 * 512 * 64;
  if (i >= total) return;
  int c = (int)(i % 64); size_t tk = i / 64;
  int pix2 = (int)(tk % (512 * 512)); int b = (int)(tk / (512 * 512));
  int y2 = pix2 / 512, x2 = pix2 % 512;
  int y = y2 >> 1, x = x2 >> 1, iy = y2 & 1, ix = x2 & 1;
  int cin = c * 4 + iy * 2 + ix;
  hs[i] = hu[((size_t)b * L0 + (size_t)y * W0 + x) * 256 + cin];
}

// ---- weight preparation ----
__global__ void k_f2bf_mat(const float* __restrict__ w, unsigned short* __restrict__ o, int n)
{ int i = blockIdx.x * blockDim.x + threadIdx.x; if (i < n) o[i] = f2bf(w[i]); }

__global__ void k_prep_taps(const float* __restrict__ w, unsigned short* __restrict__ o,
                            int Co, int Ci)
{
  int i = blockIdx.x * blockDim.x + threadIdx.x;
  int total = Co * Ci * 9;
  if (i >= total) return;
  int tap = i % 9; int ci = (i / 9) % Ci; int co = i / (9 * Ci);
  o[((size_t)tap * Ci + ci) * Co + co] = f2bf(w[(size_t)(co * Ci + ci) * 9 + tap]);
}

__global__ void k_prep_dwT(const float* __restrict__ w, float* __restrict__ o, int C)
{
  int i = blockIdx.x * blockDim.x + threadIdx.x;
  if (i >= C * 9) return;
  int tap = i % 9; int c = i / 9;
  o[tap * C + c] = w[c * 9 + tap];
}

// ---------------------------------------------------------------------------
// Host-side helpers
// ---------------------------------------------------------------------------
static void gemm(hipStream_t s, const float* A, const unsigned short* W, const float* bias,
                 const float* resid, float* C, int M, int N, int K, int act, int accum,
                 int conv, int a_nchw, int Hc, int Wc, int dy, int dx, int out_nchw)
{
  dim3 g((unsigned)((N + GBN - 1) / GBN), (unsigned)((M + GBM - 1) / GBM));
  if (conv) {
    if (a_nchw)
      k_gemm_t<1,1><<<g, 256, 0, s>>>(A, W, bias, resid, C, M, N, K, act, accum,
                                      Hc, Wc, dy, dx, out_nchw);
    else
      k_gemm_t<1,0><<<g, 256, 0, s>>>(A, W, bias, resid, C, M, N, K, act, accum,
                                      Hc, Wc, dy, dx, out_nchw);
  } else {
    k_gemm_t<0,0><<<g, 256, 0, s>>>(A, W, bias, resid, C, M, N, K, act, accum,
                                    1, 1, 0, 0, 0);
  }
}

static void convTaps(hipStream_t s, const float* A, const unsigned short* taps,
                     const float* bias, const float* resid, float* C, int M, int Cin,
                     int Cout, int Hc, int Wc, int act, int a_nchw, int out_nchw)
{
  for (int t = 0; t < 9; ++t) {
    int dy = t / 3 - 1, dx = t % 3 - 1;
    int last = (t == 8);
    gemm(s, A, taps + (size_t)t * Cin * Cout, last ? bias : nullptr,
         last ? resid : nullptr, C, M, Cout, Cin, last ? act : 0, t > 0,
         1, a_nchw, Hc, Wc, dy, dx, out_nchw);
  }
}

extern "C" void kernel_launch(void* const* d_in, const int* in_sizes, int n_in,
                              void* d_out, int out_size, void* d_ws, size_t ws_size,
                              hipStream_t stream)
{
  (void)in_sizes; (void)n_in; (void)out_size; (void)ws_size;
  auto F = [&](int i) -> const float* { return (const float*)d_in[i]; };
  const float* X = F(0);

  // input index map (dict insertion order)
  const int I_E0W=1, I_E0B=2, I_E1W=3, I_E1B=4, I_E2W=5, I_E2B=6, I_PW=7, I_PB=8;
  const int I_FUSEW=9, I_FUSEB=10;
  const int I_GB = 11;                 // groups base; 46 entries per group
  const int I_AFTER = 11 + NGROUPS*46; // 195
  const int I_UPBW = I_AFTER+1, I_UPBB = I_AFTER+2, I_UPW = I_AFTER+3;
  const int I_UPB2 = I_AFTER+4, I_LASTW = I_AFTER+5, I_LASTB = I_AFTER+6;

  // bump allocator over workspace
  size_t off = 0;
  auto allocf = [&](size_t n) -> float* {
    float* p = (float*)((char*)d_ws + off);
    off += n * sizeof(float); off = (off + 255) & ~(size_t)255; return p;
  };
  auto allocb = [&](size_t n) -> unsigned short* {
    unsigned short* p = (unsigned short*)((char*)d_ws + off);
    off += n * sizeof(unsigned short); off = (off + 255) & ~(size_t)255; return p;
  };

  const int Mtok = B0 * L0;                 // 131072
  // bf16 weight scratch (re-prepped per use)
  unsigned short* qkvbf  = allocb((size_t)D * 3 * D);
  unsigned short* projbf = allocb((size_t)D * D);
  unsigned short* fc1bf  = allocb((size_t)D * FFH);
  unsigned short* fc2bf  = allocb((size_t)FFH * D);
  unsigned short* gtaps  = allocb((size_t)9 * D * D);
  unsigned short* fusetaps  = allocb((size_t)9 * 15 * D);
  unsigned short* aftertaps = allocb((size_t)9 * D * D);
  unsigned short* upbtaps   = allocb((size_t)9 * D * MID);
  unsigned short* uptaps    = allocb((size_t)9 * MID * 256);
  unsigned short* lasttaps  = allocb((size_t)9 * MID * 3);

  // f32 buffers
  float* dwT   = allocf((size_t)9 * D);
  float* smixT = allocf((size_t)9 * FFH);
  float* z6   = allocf((size_t)B0 * 6 * L0);
  float* f0   = allocf((size_t)B0 * 32 * 128 * 128);
  float* f1   = allocf((size_t)B0 * 64 * 64 * 64);
  float* f2b  = allocf((size_t)B0 * 128 * 32 * 32);
  float* fls  = allocf((size_t)B0 * 2 * 32 * 32);
  float* flow = allocf((size_t)B0 * 2 * L0);
  float* aligned = allocf((size_t)B0 * 15 * L0);
  float* feat  = allocf((size_t)Mtok * D);
  float* bodyA = allocf((size_t)Mtok * D);
  float* bodyB = allocf((size_t)Mtok * D);
  float* gin   = allocf((size_t)Mtok * D);
  float* n1    = allocf((size_t)Mtok * D);
  float* qkvb  = allocf((size_t)Mtok * 3 * D);
  float* cf    = allocf((size_t)Mtok * D);
  float* abuf  = allocf((size_t)Mtok * D);
  float* pret  = allocf((size_t)Mtok * D);
  float* hbuf  = allocf((size_t)Mtok * FFH);
  float* h2buf = allocf((size_t)Mtok * FFH);
  float* feat2 = allocf((size_t)Mtok * D);
  float* hb    = allocf((size_t)Mtok * MID);
  float* hu    = allocf((size_t)Mtok * 256);
  float* hs    = allocf((size_t)B0 * 512 * 512 * 64);
  float* gapb  = allocf((size_t)B0 * D);
  float* cmb   = allocf((size_t)B0 * D);
  float* nq    = allocf((size_t)B0 * NH * 16);
  float* nk    = allocf((size_t)B0 * NH * 16);
  float* G     = allocf((size_t)B0 * NH * 256);
  float* attn  = allocf((size_t)B0 * NH * 256);

  auto grid1 = [](size_t n) { return (unsigned)((n + 255) / 256); };

  // ---- global conv-weight prep ----
  k_prep_taps<<<grid1(9*15*D), 256, 0, stream>>>(F(I_FUSEW), fusetaps, D, 15);
  k_prep_taps<<<grid1(9*D*D), 256, 0, stream>>>(F(I_AFTER), aftertaps, D, D);
  k_prep_taps<<<grid1(9*D*MID), 256, 0, stream>>>(F(I_UPBW), upbtaps, MID, D);
  k_prep_taps<<<grid1(9*(size_t)MID*256), 256, 0, stream>>>(F(I_UPW), uptaps, 256, MID);
  k_prep_taps<<<grid1(9*MID*3), 256, 0, stream>>>(F(I_LASTW), lasttaps, 3, MID);

  // ---- alignment: flownet + warp ----
  k_center_copy<<<grid1((size_t)B0*3*L0), 256, 0, stream>>>(X, aligned);
  const int tlist[4] = {0, 1, 3, 4};
  for (int ti = 0; ti < 4; ++ti) {
    int t = tlist[ti];
    k_concat_src_tgt<<<grid1((size_t)B0*6*L0), 256, 0, stream>>>(X, t, z6);
    k_conv_direct<<<grid1((size_t)B0*32*128*128), 256, 0, stream>>>(z6, F(I_E0W), F(I_E0B), f0, 6, 32, 256, 256, 2, 2);
    k_conv_direct<<<grid1((size_t)B0*64*64*64), 256, 0, stream>>>(f0, F(I_E1W), F(I_E1B), f1, 32, 64, 128, 128, 2, 2);
    k_conv_direct<<<grid1((size_t)B0*128*32*32), 256, 0, stream>>>(f1, F(I_E2W), F(I_E2B), f2b, 64, 128, 64, 64, 2, 2);
    k_conv_direct<<<grid1((size_t)B0*2*32*32), 256, 0, stream>>>(f2b, F(I_PW), F(I_PB), fls, 128, 2, 32, 32, 1, 0);
    k_flow_up<<<grid1((size_t)B0*2*L0), 256, 0, stream>>>(fls, flow);
    k_warp<<<grid1((size_t)B0*3*L0), 256, 0, stream>>>(X, t, flow, aligned);
  }

  // ---- fuse conv: aligned (NCHW, 15ch) -> feat (token-major [B*L][64]) ----
  convTaps(stream, aligned, fusetaps, F(I_FUSEB), nullptr, feat,
           Mtok, 15, D, H0, W0, 0, 1, 0);

  // ---- body ----
  hipMemcpyAsync(bodyA, feat, (size_t)Mtok * D * sizeof(float),
                 hipMemcpyDeviceToDevice, stream);
  float* x = bodyA; float* xalt = bodyB;

  for (int g = 0; g < NGROUPS; ++g) {
    int gbase = I_GB + g * 46;
    hipMemcpyAsync(gin, x, (size_t)Mtok * D * sizeof(float),
                   hipMemcpyDeviceToDevice, stream);
    for (int bi = 0; bi < DPG; ++bi) {
      int bb = gbase + bi * 15;
      int spatial = (bi % 2) == 0;
      // prep block weights
      k_f2bf_mat<<<grid1(D*3*D), 256, 0, stream>>>(F(bb+4), qkvbf, D*3*D);
      k_f2bf_mat<<<grid1(D*D), 256, 0, stream>>>(F(bb+5), projbf, D*D);
      k_f2bf_mat<<<grid1(D*FFH), 256, 0, stream>>>(F(bb+11), fc1bf, D*FFH);
      k_f2bf_mat<<<grid1(FFH*D), 256, 0, stream>>>(F(bb+13), fc2bf, FFH*D);
      k_prep_dwT<<<grid1(D*9), 256, 0, stream>>>(F(bb+7), dwT, D);
      k_prep_dwT<<<grid1(FFH*9), 256, 0, stream>>>(F(bb+12), smixT, FFH);

      // n1 = LN(x); cf = gelu(dwconv(n1)); qkv = n1 @ qkv_w
      k_layernorm<<<grid1(Mtok), 256, 0, stream>>>(x, F(bb+0), F(bb+1), n1, Mtok);
      k_dwconv4<<<grid1((size_t)Mtok*(D/4)), 256, 0, stream>>>(n1, dwT, cf, D, 1);
      gemm(stream, n1, qkvbf, nullptr, nullptr, qkvb, Mtok, 3*D, D,
           0, 0, 0, 0, 0, 0, 0, 0, 0);

      if (spatial) {
        k_win_attn<<<B0*(H0/8)*(W0/8), 128, 0, stream>>>(qkvb, F(bb+14), pret);
        gemm(stream, pret, projbf, F(bb+6), nullptr, abuf, Mtok, D, D,
             0, 0, 0, 0, 0, 0, 0, 0, 0);
        k_gap<<<B0*D, 256, 0, stream>>>(cf, gapb);
        k_cm<<<B0, 64, 0, stream>>>(gapb, F(bb+9), F(bb+10), cmb);
        k_spat_fuse<<<grid1((size_t)Mtok*(D/4)), 256, 0, stream>>>(x, abuf, cf, cmb);
      } else {
        k_chan_norms<<<B0*NH*16, 256, 0, stream>>>(qkvb, nq, nk);
        k_zero<<<grid1(B0*NH*256), 256, 0, stream>>>(G, B0*NH*256);
        k_chan_gram<<<B0*NH*64, 256, 0, stream>>>(qkvb, G);
        k_chan_softmax<<<B0*NH, 16, 0, stream>>>(G, nq, nk, F(bb+14), attn);
        k_chan_apply<<<grid1((size_t)Mtok*NH), 256, 0, stream>>>(qkvb, attn, pret);
        gemm(stream, pret, projbf, F(bb+6), nullptr, abuf, Mtok, D, D,
             0, 0, 0, 0, 0, 0, 0, 0, 0);
        k_chan_fuse<<<grid1(Mtok), 256, 0, stream>>>(x, abuf, cf, F(bb+8), Mtok);
      }

      // MLP: n2 -> gelu(fc1) -> smix dwconv -> fc2 (+x)
      k_layernorm<<<grid1(Mtok), 256, 0, stream>>>(x, F(bb+2), F(bb+3), n1, Mtok);
      gemm(stream, n1, fc1bf, nullptr, nullptr, hbuf, Mtok, FFH, D,
           1, 0, 0, 0, 0, 0, 0, 0, 0);
      k_dwconv4<<<grid1((size_t)Mtok*(FFH/4)), 256, 0, stream>>>(hbuf, smixT, h2buf, FFH, 0);
      gemm(stream, h2buf, fc2bf, nullptr, x, x, Mtok, D, FFH,
           0, 0, 0, 0, 0, 0, 0, 0, 0);
    }
    // group conv + residual(group input)
    k_prep_taps<<<grid1(9*D*D), 256, 0, stream>>>(F(gbase+45), gtaps, D, D);
    convTaps(stream, x, gtaps, nullptr, gin, xalt, Mtok, D, D, H0, W0, 0, 0, 0);
    float* tmpp = x; x = xalt; xalt = tmpp;
  }

  // ---- tail: after conv + feat residual, upsample, last conv ----
  convTaps(stream, x, aftertaps, nullptr, feat, feat2, Mtok, D, D, H0, W0, 0, 0, 0);
  convTaps(stream, feat2, upbtaps, F(I_UPBB), nullptr, hb, Mtok, D, MID, H0, W0, 2, 0, 0);
  convTaps(stream, hb, uptaps, F(I_UPB2), nullptr, hu, Mtok, MID, 256, H0, W0, 0, 0, 0);
  k_pixshuf<<<grid1((size_t)B0*512*512*64), 256, 0, stream>>>(hu, hs);
  convTaps(stream, hs, lasttaps, F(I_LASTB), nullptr, (float*)d_out,
           B0*512*512, MID, 3, 512, 512, 0, 0, 1);
}
